// MHA_41480794144945
// MI455X (gfx1250) — compile-verified, matrix operand order checked by reading, not run on hardware
//
#include <hip/hip_runtime.h>
#include <hip/hip_bf16.h>

// Problem constants (from reference): B=2, T=2048, D=1024, H=16, DK=64
#define BB  2
#define TT  2048
#define DD  1024
#define HH  16
#define DKK 64

typedef __attribute__((ext_vector_type(16))) _Float16 f16x16;
typedef __attribute__((ext_vector_type(8)))  _Float16 f16x8;
typedef __attribute__((ext_vector_type(8)))  float    f32x8;
typedef __attribute__((ext_vector_type(4)))  float    f32x4;

// ---------------------------------------------------------------------------
// WMMA helper: D = A(16x32 f16) * B(32x16 f16) + C(16x16 f32)
// ---------------------------------------------------------------------------
__device__ inline f32x8 wmma_f16(f16x16 a, f16x16 b, f32x8 c) {
  return __builtin_amdgcn_wmma_f32_16x16x32_f16(
      /*neg_a=*/false, a, /*neg_b=*/false, b,
      /*c_mod=*/(short)0, c, /*reuse_a=*/false, /*reuse_b=*/false);
}

// ---------------------------------------------------------------------------
// Load one 16x32 f16 operand (A: lane = row; B: lane = column, mirrored
// layout). Lane L (m = L&15, kh = L>>4) holds K in [kh*8, kh*8+8) U
// [16+kh*8, 16+kh*8+8): two contiguous 16B chunks. Generic ptr (global/LDS).
// ---------------------------------------------------------------------------
__device__ inline f16x16 load_tile_ab(const _Float16* p, int rstride) {
  const int lane = threadIdx.x & 31;
  const _Float16* q = p + (long)(lane & 15) * rstride + (lane >> 4) * 8;
  f16x8 lo = *(const f16x8*)(q);
  f16x8 hi = *(const f16x8*)(q + 16);
  f16x16 r;
#pragma unroll
  for (int i = 0; i < 8; ++i) { r[i] = lo[i]; r[8 + i] = hi[i]; }
  return r;
}

// ---------------------------------------------------------------------------
// CDNA5 async global->LDS copy (16 bytes per lane), tracked by ASYNCcnt.
// Generic pointers to __shared__ carry the wave-relative LDS byte offset in
// their low 32 bits (flat aperture mapping), which is what VDST wants.
// ---------------------------------------------------------------------------
__device__ inline unsigned lds_addr_of(const void* p) {
  return (unsigned)(unsigned long long)(size_t)p;
}
__device__ inline void async_copy16(unsigned lds_byte, unsigned long long g) {
  asm volatile("global_load_async_to_lds_b128 %0, %1, off"
               :: "v"(lds_byte), "v"(g) : "memory");
}
__device__ inline void wait_async_le16() {
  asm volatile("s_wait_asynccnt 0x10" ::: "memory");
}
__device__ inline void wait_async_0() {
  asm volatile("s_wait_asynccnt 0x0" ::: "memory");
}

// Stage a contiguous 32x64 f16 K tile (4 KB) into LDS: 8 x B128 per lane.
__device__ inline void stage_k(const _Float16* gsrc, unsigned lds_base) {
  const int lane = threadIdx.x & 31;
  unsigned long long g = (unsigned long long)(size_t)gsrc + (unsigned)(lane * 16);
  unsigned l = lds_base + lane * 16;
#pragma unroll
  for (int i = 0; i < 8; ++i) async_copy16(l + i * 512, g + i * 512);
}

// Stage a 64(dk) x 32(key) f16 V tile (rows of 64B, global stride T*2 bytes)
// into LDS laid out [dk][32 keys] (row stride 64B): 8 x B128 per lane.
__device__ inline void stage_v(const _Float16* gsrc, unsigned lds_base) {
  const int lane = threadIdx.x & 31;
  const int sub = lane & 3;        // 16B chunk within a 64B row
  const int rb  = lane >> 2;       // row 0..7 within each group of 8
#pragma unroll
  for (int i = 0; i < 8; ++i) {
    const int r = i * 8 + rb;      // dk row 0..63
    unsigned long long g =
        (unsigned long long)(size_t)gsrc + (unsigned long long)r * (TT * 2) + sub * 16;
    async_copy16(lds_base + r * 64 + sub * 16, g);
  }
}

// ---------------------------------------------------------------------------
// Tiled GEMM: Out = X @ W^T + bias, (M x 1024) x (1024 x 1024).
// Block tile 64(M) x 128(N), 256 threads = 8 waves (2x4), 32x32 per wave.
// ---------------------------------------------------------------------------
template <bool IN_F16, int OUT_MODE>
__global__ __launch_bounds__(256) void gemm_wmma_kern(
    const void* __restrict__ Xv, const float* __restrict__ W,
    const float* __restrict__ bias, void* __restrict__ Out) {
  __shared__ __align__(16) _Float16 As[64 * 32];
  __shared__ __align__(16) _Float16 Bs[128 * 32];

  const int tid   = threadIdx.x;
  const int lane  = tid & 31;
  const int wave  = tid >> 5;
  const int waveM = wave >> 2;   // 0..1
  const int waveN = wave & 3;    // 0..3
  const int tileN = blockIdx.x;  // 1024/128 = 8
  const int tileM = blockIdx.y;  // M/64

  f32x8 acc[2][2];
#pragma unroll
  for (int i = 0; i < 2; ++i)
#pragma unroll
    for (int j = 0; j < 2; ++j) acc[i][j] = f32x8{};

  const int arow = tid >> 2;         // 0..63
  const int acol = (tid & 3) * 8;    // 0,8,16,24
  const int brow = tid >> 1;         // 0..127
  const int bcol = (tid & 1) * 16;   // 0,16

  const float*    Xf = (const float*)Xv;
  const _Float16* Xh = (const _Float16*)Xv;
  const unsigned  as_lds = lds_addr_of(As);

  for (int kk = 0; kk < DD; kk += 32) {
    // ---- stage A (64x32) into LDS as f16 ----
    {
      long gro = (long)(tileM * 64 + arow) * DD + kk + acol;
      if constexpr (IN_F16) {
        // f16 input: DMA straight into LDS via the async path.
        async_copy16(as_lds + (unsigned)(arow * 64 + acol * 2),
                     (unsigned long long)(size_t)(Xh + gro));
      } else {
        f32x4 x0 = *(const f32x4*)(Xf + gro);
        f32x4 x1 = *(const f32x4*)(Xf + gro + 4);
        f16x8 va;
#pragma unroll
        for (int i = 0; i < 4; ++i) {
          va[i] = (_Float16)x0[i];
          va[4 + i] = (_Float16)x1[i];
        }
        *(f16x8*)(As + arow * 32 + acol) = va;
      }
    }
    // ---- stage B (128x32) = rows of W (W[j][d]) into LDS as f16 ----
    {
      long gro = (long)(tileN * 128 + brow) * DD + kk + bcol;
      if (kk + 32 < DD) __builtin_prefetch(W + gro + 32, 0, 1);
      f32x4 w0 = *(const f32x4*)(W + gro);
      f32x4 w1 = *(const f32x4*)(W + gro + 4);
      f32x4 w2 = *(const f32x4*)(W + gro + 8);
      f32x4 w3 = *(const f32x4*)(W + gro + 12);
      f16x8 v0, v1;
#pragma unroll
      for (int i = 0; i < 4; ++i) {
        v0[i] = (_Float16)w0[i]; v0[4 + i] = (_Float16)w1[i];
        v1[i] = (_Float16)w2[i]; v1[4 + i] = (_Float16)w3[i];
      }
      *(f16x8*)(Bs + brow * 32 + bcol)     = v0;
      *(f16x8*)(Bs + brow * 32 + bcol + 8) = v1;
    }
    if constexpr (IN_F16) wait_async_0();
    __syncthreads();

    f16x16 a0 = load_tile_ab(As + (waveM * 32 + 0)  * 32, 32);
    f16x16 a1 = load_tile_ab(As + (waveM * 32 + 16) * 32, 32);
    f16x16 b0 = load_tile_ab(Bs + (waveN * 32 + 0)  * 32, 32);
    f16x16 b1 = load_tile_ab(Bs + (waveN * 32 + 16) * 32, 32);
    acc[0][0] = wmma_f16(a0, b0, acc[0][0]);
    acc[0][1] = wmma_f16(a0, b1, acc[0][1]);
    acc[1][0] = wmma_f16(a1, b0, acc[1][0]);
    acc[1][1] = wmma_f16(a1, b1, acc[1][1]);
    __syncthreads();
  }

  // ---- epilogue: bias + store ----
  const int half = lane >> 4;
  const int n    = lane & 15;
#pragma unroll
  for (int i = 0; i < 2; ++i) {
#pragma unroll
    for (int j = 0; j < 2; ++j) {
      const int gn = tileN * 128 + waveN * 32 + j * 16 + n;
      const float bvv = bias[gn];
#pragma unroll
      for (int r = 0; r < 8; ++r) {
        const int gm = tileM * 64 + waveM * 32 + i * 16 + r + half * 8;
        const float val = acc[i][j][r] + bvv;
        if constexpr (OUT_MODE == 0) {
          const int b = gm >> 11, t = gm & (TT - 1);
          const int h = gn >> 6,  dk = gn & 63;
          ((_Float16*)Out)[(((long)(b * HH + h) * TT + t) * DKK) + dk] =
              (_Float16)val;
        } else if constexpr (OUT_MODE == 1) {
          const int b = gm >> 11, t = gm & (TT - 1);
          const int h = gn >> 6,  dk = gn & 63;
          ((_Float16*)Out)[(((long)(b * HH + h) * DKK + dk) * TT) + t] =
              (_Float16)val;
        } else {
          ((float*)Out)[(long)gm * DD + gn] = val;
        }
      }
    }
  }
}

// ---------------------------------------------------------------------------
// Flash attention: one wave per (b, h, 32-query tile). Keys stream in chunks
// of 32, double-buffered in LDS via GLOBAL_LOAD_ASYNC_TO_LDS_B128 and
// S_WAIT_ASYNCCNT (async loads retire in order, so waiting <=16 outstanding
// guarantees the current buffer while next-chunk copies stay in flight).
//   qh: [B,H,T,DK] f16, kh: [B,H,T,DK] f16, vt: [B,H,DK,T] f16
//   ctx out: [B*T, D] f16 (row = b*T+t, col = h*64+dk)
// ---------------------------------------------------------------------------
__global__ __launch_bounds__(32) void flash_attn_kern(
    const _Float16* __restrict__ qh, const _Float16* __restrict__ kh,
    const _Float16* __restrict__ vt, const int* __restrict__ maskp,
    const int* __restrict__ causalp, _Float16* __restrict__ ctx) {
  __shared__ __align__(16) _Float16 Kl[2][32 * DKK];  // [key][dk]
  __shared__ __align__(16) _Float16 Vl[2][DKK * 32];  // [dk][key]
  __shared__ __align__(16) _Float16 Pl[32 * 32];      // [qrow][key]

  const int lane = threadIdx.x & 31;
  const int half = lane >> 4;
  const int n    = lane & 15;
  const int qt = blockIdx.x;   // T/32
  const int h  = blockIdx.y;   // H
  const int b  = blockIdx.z;   // B
  const int q0 = qt * 32;
  const int causal = *causalp;

  const long bh = (long)b * HH + h;
  const _Float16* Q  = qh + (bh * TT + q0) * DKK;
  const _Float16* Kg = kh + bh * TT * DKK;
  const _Float16* Vg = vt + bh * DKK * TT;

  // A operands: 32 q-rows x 64 dk -> 2 row-tiles x 2 K-chunks
  f16x16 aq[2][2];
#pragma unroll
  for (int mi = 0; mi < 2; ++mi) {
    aq[mi][0] = load_tile_ab(Q + mi * 16 * DKK, DKK);
    aq[mi][1] = load_tile_ab(Q + mi * 16 * DKK + 32, DKK);
  }

  f32x8 O[2][4];
#pragma unroll
  for (int mi = 0; mi < 2; ++mi)
#pragma unroll
    for (int j = 0; j < 4; ++j) O[mi][j] = f32x8{};
  float mrow[2][8], lrow[2][8];
#pragma unroll
  for (int mi = 0; mi < 2; ++mi)
#pragma unroll
    for (int r = 0; r < 8; ++r) { mrow[mi][r] = -1e30f; lrow[mi][r] = 0.f; }

  const unsigned klb[2] = {lds_addr_of(&Kl[0][0]), lds_addr_of(&Kl[1][0])};
  const unsigned vlb[2] = {lds_addr_of(&Vl[0][0]), lds_addr_of(&Vl[1][0])};

  const int kend = causal ? (q0 + 32) : TT;  // multiple of 32
  stage_k(Kg, klb[0]);
  stage_v(Vg, vlb[0]);

  for (int kc = 0; kc < kend; kc += 32) {
    const int p = (kc >> 5) & 1;
    if (kc + 32 < kend) {
      stage_k(Kg + (long)(kc + 32) * DKK, klb[p ^ 1]);
      stage_v(Vg + (kc + 32), vlb[p ^ 1]);
      wait_async_le16();  // current buffer complete, next stays in flight
    } else {
      wait_async_0();
    }
    const _Float16* Kc = &Kl[p][0];
    const _Float16* Vc = &Vl[p][0];

    // ---- S = Q K^T : 32 q-rows x 32 keys, interleaved for hazard cover ----
    f16x16 bk00 = load_tile_ab(Kc, DKK);
    f16x16 bk01 = load_tile_ab(Kc + 32, DKK);
    f16x16 bk10 = load_tile_ab(Kc + 16 * DKK, DKK);
    f16x16 bk11 = load_tile_ab(Kc + 16 * DKK + 32, DKK);
    f32x8 s[2][2];
    s[0][0] = wmma_f16(aq[0][0], bk00, f32x8{});
    s[0][1] = wmma_f16(aq[0][0], bk10, f32x8{});
    s[1][0] = wmma_f16(aq[1][0], bk00, f32x8{});
    s[1][1] = wmma_f16(aq[1][0], bk10, f32x8{});
    s[0][0] = wmma_f16(aq[0][1], bk01, s[0][0]);
    s[0][1] = wmma_f16(aq[0][1], bk11, s[0][1]);
    s[1][0] = wmma_f16(aq[1][1], bk01, s[1][0]);
    s[1][1] = wmma_f16(aq[1][1], bk11, s[1][1]);

    // ---- scale + masks + online softmax (rows live in 16-lane halves) ----
    const float NEG = -1e30f;
    const int key0 = kc + n;
    const int key1 = kc + 16 + n;
    const int mk0 = maskp[(long)b * TT + key0];
    const int mk1 = maskp[(long)b * TT + key1];
    float alpha[2][8], mnew[2][8];
#pragma unroll
    for (int mi = 0; mi < 2; ++mi) {
#pragma unroll
      for (int r = 0; r < 8; ++r) {
        const int qrow = q0 + mi * 16 + r + half * 8;
        float v0 = s[mi][0][r] * 0.125f;  // 1/sqrt(64)
        float v1 = s[mi][1][r] * 0.125f;
        if ((causal && key0 > qrow) || mk0 == 0) v0 = NEG;
        if ((causal && key1 > qrow) || mk1 == 0) v1 = NEG;
        s[mi][0][r] = v0; s[mi][1][r] = v1;
      }
#pragma unroll
      for (int r = 0; r < 8; ++r) {
        float mx = fmaxf(s[mi][0][r], s[mi][1][r]);
        for (int off = 1; off < 16; off <<= 1)
          mx = fmaxf(mx, __shfl_xor(mx, off, 32));
        mnew[mi][r]  = fmaxf(mrow[mi][r], mx);
        alpha[mi][r] = __expf(mrow[mi][r] - mnew[mi][r]);
        mrow[mi][r]  = mnew[mi][r];
      }
#pragma unroll
      for (int r = 0; r < 8; ++r) {
        const float p0 =
            (s[mi][0][r] < -1e29f) ? 0.f : __expf(s[mi][0][r] - mnew[mi][r]);
        const float p1 =
            (s[mi][1][r] < -1e29f) ? 0.f : __expf(s[mi][1][r] - mnew[mi][r]);
        float rs = p0 + p1;
        for (int off = 1; off < 16; off <<= 1) rs += __shfl_xor(rs, off, 32);
        lrow[mi][r] = lrow[mi][r] * alpha[mi][r] + rs;
        // transpose P through LDS: C layout -> row-major [32][32]
        const int m = mi * 16 + r + half * 8;
        Pl[m * 32 + n]      = (_Float16)p0;
        Pl[m * 32 + 16 + n] = (_Float16)p1;
      }
#pragma unroll
      for (int j = 0; j < 4; ++j)
#pragma unroll
        for (int r = 0; r < 8; ++r) O[mi][j][r] *= alpha[mi][r];
    }

    // ---- O += P(32x32) @ V(32x64) (same-wave DS ops are in order) ----
    const f16x16 ap0 = load_tile_ab(Pl, 32);
    const f16x16 ap1 = load_tile_ab(Pl + 16 * 32, 32);
#pragma unroll
    for (int j = 0; j < 4; ++j) {
      f16x16 bvt = load_tile_ab(Vc + j * 16 * 32, 32);
      O[0][j] = wmma_f16(ap0, bvt, O[0][j]);
      O[1][j] = wmma_f16(ap1, bvt, O[1][j]);
    }
  }

  // ---- normalize + write context in flat [B*T, D] f16 ----
#pragma unroll
  for (int mi = 0; mi < 2; ++mi) {
#pragma unroll
    for (int r = 0; r < 8; ++r) {
      const float inv = 1.0f / lrow[mi][r];
      const int t = q0 + mi * 16 + r + half * 8;
      const long rowbase = ((long)b * TT + t) * DD + h * DKK;
#pragma unroll
      for (int j = 0; j < 4; ++j)
        ctx[rowbase + j * 16 + n] = (_Float16)(O[mi][j][r] * inv);
    }
  }
}

// ---------------------------------------------------------------------------
// Launch: 3 projection GEMMs -> flash attention -> output GEMM.
// Workspace: qh | kh | vt | ctx, each B*T*D f16 = 8 MB (32 MB total).
// ---------------------------------------------------------------------------
extern "C" void kernel_launch(void* const* d_in, const int* in_sizes, int n_in,
                              void* d_out, int out_size, void* d_ws,
                              size_t ws_size, hipStream_t stream) {
  const float* q   = (const float*)d_in[0];
  const float* k   = (const float*)d_in[1];
  const float* v   = (const float*)d_in[2];
  const float* Wq  = (const float*)d_in[3];
  const float* bq  = (const float*)d_in[4];
  const float* Wk  = (const float*)d_in[5];
  const float* bk  = (const float*)d_in[6];
  const float* Wv  = (const float*)d_in[7];
  const float* bv  = (const float*)d_in[8];
  const float* Wo  = (const float*)d_in[9];
  const float* bo  = (const float*)d_in[10];
  const int* mask  = (const int*)d_in[11];
  const int* caus  = (const int*)d_in[12];

  const size_t elems = (size_t)BB * TT * DD;
  _Float16* qh  = (_Float16*)d_ws;
  _Float16* khd = qh + elems;
  _Float16* vtd = khd + elems;
  _Float16* ctx = vtd + elems;

  dim3 gg(DD / 128, (BB * TT) / 64);
  gemm_wmma_kern<false, 0><<<gg, 256, 0, stream>>>(q, Wq, bq, qh);
  gemm_wmma_kern<false, 0><<<gg, 256, 0, stream>>>(k, Wk, bk, khd);
  gemm_wmma_kern<false, 1><<<gg, 256, 0, stream>>>(v, Wv, bv, vtd);
  flash_attn_kern<<<dim3(TT / 32, HH, BB), 32, 0, stream>>>(qh, khd, vtd, mask,
                                                            caus, ctx);
  gemm_wmma_kern<true, 2><<<gg, 256, 0, stream>>>(ctx, Wo, bo, (float*)d_out);
}
